// CRF_16217796510032
// MI455X (gfx1250) — matrix operand order repaired and checked by reading, not compiled, not run on hardware
//
#include <hip/hip_runtime.h>

typedef __attribute__((ext_vector_type(16))) _Float16 v16h;
typedef __attribute__((ext_vector_type(8)))  float    v8f;
typedef __attribute__((ext_vector_type(4)))  unsigned int v4u;

#define CRF_B 128
#define CRF_S 512
#define CRF_T 256
#define ROWS  16    // batch rows per workgroup
#define WAVES 16    // one 16-wide column tile per wave (16*16 = 256 = T)
#define PADH  264   // padded f16 elements per expA row (528B stride -> conflict-free b128 reads)

// DPP16 ROW_XMASK butterfly (lane i <-> lane i^mask within each 16-lane row).
// bound_ctrl=1 zero-fills invalid lanes.
// Max done as UNSIGNED INT max: valid because all values are non-negative finite
// floats (exp-space), where IEEE order == unsigned int order; avoids the
// v_max_num canonicalization self-maxes.
template <int CTRL>
__device__ __forceinline__ unsigned dppMaxU(unsigned v) {
  unsigned m = (unsigned)__builtin_amdgcn_update_dpp(0, (int)v, CTRL, 0xF, 0xF, true);
  return v > m ? v : m;          // v_max_u32
}
template <int CTRL>
__device__ __forceinline__ float dppAdd(float v) {
  int m = __builtin_amdgcn_update_dpp(0, __builtin_bit_cast(int, v), CTRL, 0xF, 0xF, true);
  return v + __builtin_bit_cast(float, m);
}
#define XM1 0x161
#define XM2 0x162
#define XM4 0x164
#define XM8 0x168

__device__ __forceinline__ unsigned redMaxU16(unsigned v) {
  v = dppMaxU<XM1>(v); v = dppMaxU<XM2>(v);
  v = dppMaxU<XM4>(v); v = dppMaxU<XM8>(v);
  return v;
}
__device__ __forceinline__ float redAdd16(float v) {
  v = dppAdd<XM1>(v); v = dppAdd<XM2>(v);
  v = dppAdd<XM4>(v); v = dppAdd<XM8>(v);
  return v;
}

struct SMem {
  _Float16 expA[2][ROWS][PADH];       // exp(alpha - m_row), double buffered, max per row == 1
  float    rinv[ROWS];                // 1/rowM for current rescale step
  unsigned wavemax[ROWS][WAVES];      // per-wave partial row maxes (bitcast f32, transposed)
  float    wavesum[ROWS][WAVES];      // per-wave partial row sums (finalize)
  unsigned short maskbits[CRF_S];     // bit r = mask[bbase+r][t] (0xFFFF for partition block)
};

__global__ __launch_bounds__(512, 1)
void crf_forward(const float* __restrict__ em,
                 const unsigned char* __restrict__ mask,
                 const float* __restrict__ startT,
                 const float* __restrict__ endT,
                 const float* __restrict__ trans,
                 float* __restrict__ ws)
{
  __shared__ SMem sm;
  const int tid   = threadIdx.x;
  const int wave  = tid >> 5;
  const int lane  = tid & 31;
  const int half  = lane >> 4;
  const int l16   = lane & 15;
  const int bbase = blockIdx.x * ROWS;
  const bool use_mask = (blockIdx.y == 0);   // y=0: masked "score", y=1: "partition"
  const int col   = (wave << 4) + l16;

  const size_t rowStride = (size_t)CRF_S * CRF_T;
  const float* emBase = em + (size_t)bbase * rowStride + col;
  const unsigned char* mkBase = mask + (size_t)bbase * CRF_S;

  // ---- pack this block's mask tile into bits: maskbits[t] bit r = mask[r][t] ----
  {
    unsigned m = 0xFFFFu;                    // partition block: mask ignored
    if (use_mask) {
      m = 0u;
      #pragma unroll
      for (int r = 0; r < ROWS; ++r)
        m |= (unsigned)(mkBase[(size_t)r * CRF_S + tid] != 0) << r;
    }
    sm.maskbits[tid] = (unsigned short)m;    // tid in [0,512) == CRF_S
  }

  // ---- B fragments: exp(transitions[k][col]) in f16, resident in VGPRs ----
  v16h bfrag[8];
  #pragma unroll
  for (int s = 0; s < 8; ++s) {
    const int kb = s * 32 + half * 16;
    v16h f;
    #pragma unroll
    for (int h = 0; h < 16; ++h)
      f[h] = (_Float16)__expf(trans[(size_t)(kb + h) * CRF_T + col]);
    bfrag[s] = f;
  }

  // Lane state (C-fragment layout: row = half*8 + k, column = col):
  float ea[8];          // published expA value for (row, col)
  float mrow = 0.f;     // row offset m for row 'wave' (meaningful on lane 0 of each wave)

  // ---- t = 0 init in exp-space: y = exp(start + em0), m starts at 0 ----
  float y[8];
  const float sc = startT[col];
  #pragma unroll
  for (int k = 0; k < 8; ++k)
    y[k] = __expf(sc + emBase[(size_t)(half * 8 + k) * rowStride]);

  int pr = 0;
  // ---- publish (init): row maxes -> inv -> rescale ----
  #pragma unroll
  for (int k = 0; k < 8; ++k) {
    unsigned rm = redMaxU16(__builtin_bit_cast(unsigned, y[k]));
    if (l16 == 0) sm.wavemax[half * 8 + k][wave] = rm;
  }
  __syncthreads();
  {
    unsigned x = redMaxU16(sm.wavemax[wave][l16]);   // 16 partials for row 'wave'
    if (lane == 0) {
      const float inv = __builtin_amdgcn_rcpf(__builtin_bit_cast(float, x));
      sm.rinv[wave] = inv;
      mrow -= __logf(inv);                           // m' = m + log(rowM)
    }
  }
  __syncthreads();
  #pragma unroll
  for (int k = 0; k < 8; ++k) {
    const int row = half * 8 + k;
    ea[k] = y[k] * sm.rinv[row];
    sm.expA[0][row][col] = (_Float16)ea[k];
  }
  __syncthreads();

  for (int t = 1; t < CRF_S; ++t) {
    // ---- issue all global loads FIRST so latency overlaps the matmul ----
    const float* emt = emBase + (size_t)t * CRF_T;
    float ex[8];
    #pragma unroll
    for (int k = 0; k < 8; ++k)
      ex[k] = emt[(size_t)(half * 8 + k) * rowStride];
    const unsigned mb = sm.maskbits[t];        // broadcast ds load, hidden under matmul
    if (t + 1 < CRF_S) {
      #pragma unroll
      for (int k = 0; k < 8; ++k)
        __builtin_prefetch(emt + CRF_T + (size_t)(half * 8 + k) * rowStride, 0, 3);
    }

    // ---- GEMM in exp-space: c[row][col] = sum_k expA[row][k] * expT[k][col] ----
    v8f c0 = {}, c1 = {};
    const _Float16* arow = sm.expA[pr][l16];
    const int aoff = half * 16;
    #pragma unroll
    for (int s = 0; s < 8; s += 2) {
      v16h a0, a1;
      ((v4u*)&a0)[0] = *(const v4u*)(arow + s * 32 + aoff);
      ((v4u*)&a0)[1] = *(const v4u*)(arow + s * 32 + 8 + aoff);
      ((v4u*)&a1)[0] = *(const v4u*)(arow + (s + 1) * 32 + aoff);
      ((v4u*)&a1)[1] = *(const v4u*)(arow + (s + 1) * 32 + 8 + aoff);
      c0 = __builtin_amdgcn_wmma_f32_16x16x32_f16(false, a0, false, bfrag[s],
                                                  (short)0, c0, false, false);
      c1 = __builtin_amdgcn_wmma_f32_16x16x32_f16(false, a1, false, bfrag[s + 1],
                                                  (short)0, c1, false, false);
    }

    // ---- y = c * exp(em); masked rows keep previous expA (exact: rowM stays 1) ----
    #pragma unroll
    for (int k = 0; k < 8; ++k) {
      const int row = half * 8 + k;
      const float yn = (c0[k] + c1[k]) * __expf(ex[k]);
      y[k] = ((mb >> row) & 1u) ? yn : ea[k];  // branchless v_cndmask
    }

    // ---- publish: row maxes -> inv -> rescale, swap buffers ----
    const int pw = pr ^ 1;
    #pragma unroll
    for (int k = 0; k < 8; ++k) {
      unsigned rm = redMaxU16(__builtin_bit_cast(unsigned, y[k]));
      if (l16 == 0) sm.wavemax[half * 8 + k][wave] = rm;
    }
    __syncthreads();
    {
      unsigned x = redMaxU16(sm.wavemax[wave][l16]);
      if (lane == 0) {
        const float inv = __builtin_amdgcn_rcpf(__builtin_bit_cast(float, x));
        sm.rinv[wave] = inv;
        mrow -= __logf(inv);
      }
    }
    __syncthreads();
    #pragma unroll
    for (int k = 0; k < 8; ++k) {
      const int row = half * 8 + k;
      ea[k] = y[k] * sm.rinv[row];
      sm.expA[pw][row][col] = (_Float16)ea[k];
    }
    __syncthreads();
    pr = pw;
  }

  // ---- finalize: lse_j(alpha + end) = m + log( sum_j expA_j * exp(end_j) ) ----
  const float eend = __expf(endT[col]);
  #pragma unroll
  for (int k = 0; k < 8; ++k) {
    float e = redAdd16(ea[k] * eend);
    if (l16 == 0) sm.wavesum[half * 8 + k][wave] = e;
  }
  __syncthreads();
  {
    float s = redAdd16(sm.wavesum[wave][l16]);
    if (lane == 0)
      ws[blockIdx.y * CRF_B + bbase + wave] = mrow + __logf(s);
  }
}

__global__ void crf_diff(const float* __restrict__ ws, float* __restrict__ out) {
  const int b = blockIdx.x * blockDim.x + threadIdx.x;
  if (b < CRF_B) out[b] = ws[CRF_B + b] - ws[b];   // partition - score
}

extern "C" void kernel_launch(void* const* d_in, const int* in_sizes, int n_in,
                              void* d_out, int out_size, void* d_ws, size_t ws_size,
                              hipStream_t stream) {
  const float*         em = (const float*)d_in[0];
  const unsigned char* mk = (const unsigned char*)d_in[1];
  const float*         st = (const float*)d_in[2];
  const float*         en = (const float*)d_in[3];
  const float*         tr = (const float*)d_in[4];
  float* ws  = (float*)d_ws;
  float* out = (float*)d_out;

  dim3 grid(CRF_B / ROWS, 2);   // 8 batch tiles x {score, partition}
  crf_forward<<<grid, ROWS * WAVES * 2, 0, stream>>>(em, mk, st, en, tr, ws);
  crf_diff<<<1, CRF_B, 0, stream>>>(ws, out);
}
